// MSFusion_81552839016902
// MI455X (gfx1250) — compile-verified, matrix operand order checked by reading, not run on hardware
//
#include <hip/hip_runtime.h>
#include <hip/hip_bf16.h>

typedef __attribute__((ext_vector_type(16))) __bf16 v16bf;
typedef __attribute__((ext_vector_type(2)))  __bf16 v2bf;
typedef __attribute__((ext_vector_type(8)))  float  v8f;

#define TT   2048
#define BB   4
#define DMOD 256
#define DM2  512
#define DIN  1024
#define NH   16
#define HD   64
#define DST  128
#define DPRJ 2320
#define CCH  1280
#define MROWS (BB * TT)   // 8192
#define STR  20           // padded LDS row stride (dwords): 16B-aligned, bank-spread

__device__ __forceinline__ float sigm(float x) { return 1.0f / (1.0f + __expf(-x)); }
__device__ __forceinline__ float siluf(float x) { return x * sigm(x); }

__device__ __forceinline__ unsigned int pack_bf16x2(float lo, float hi) {
  v2bf p;
  p.x = (__bf16)lo;
  p.y = (__bf16)hi;
  return __builtin_bit_cast(unsigned int, p);
}

union FragU {
  v16bf v;
  unsigned int u[8];
};

// ---------------------------------------------------------------------------
// Fragment builders from packed-dword LDS tiles (both tiles lane-major with
// padded stride STR, so each lane's fragment = two runs of 4 consecutive
// dwords -> 2x ds_load_b128, no register shuffling).
// CDNA5 16-bit A 16x32 layout: lane m=lane&15, khalf=(lane>>4)*8;
// vgpr v holds k = (v<4 ? 2v : 16+2(v-4)) + khalf  (pairs -> one dword).
// ---------------------------------------------------------------------------
__device__ __forceinline__ v16bf frag_ld(const unsigned int* __restrict__ S,
                                         int lanebase, int lane) {
  const int m  = lanebase + (lane & 15);
  const int kh = (lane >> 4) << 3;
  FragU f;
#pragma unroll
  for (int v = 0; v < 8; ++v) {
    const int kb = ((v < 4) ? (2 * v) : (16 + 2 * (v - 4))) + kh;
    f.u[v] = S[m * STR + (kb >> 1)];
  }
  return f.v;
}

// ---------------------------------------------------------------------------
// bf16 WMMA GEMM: C(MxN) = A(MxK) * B(KxN) (+bias).  A/B are bf16 in memory.
// Block tile 128x64, 8 waves, each wave owns 32x32 (4 WMMAs per k-step).
// ---------------------------------------------------------------------------
enum { A_NORMAL = 0, A_FLIP = 1, A_XGATHER = 2 };
enum { B_NORMAL = 0, B_TRANS = 1 };
enum { EPI_STORE = 0, EPI_BIAS = 1, EPI_FLIPROW = 2 };

template <int AMODE, int BMODE, int EPI, int DUAL>
__global__ __launch_bounds__(256) void gemm_wmma(
    const __bf16* __restrict__ A, const __bf16* __restrict__ B,
    const float* __restrict__ bias, float* __restrict__ C,
    __bf16* __restrict__ Cbf,
    int M, int N, int K, int lda, int ldb, int ldc,
    long astride, long bstride, long cstride) {
  A += (long)blockIdx.z * astride;
  B += (long)blockIdx.z * bstride;
  C += (long)blockIdx.z * cstride;

  __shared__ __align__(16) unsigned int As[128 * STR];  // 10 KB, rows m-major
  __shared__ __align__(16) unsigned int Bs[64 * STR];   //  5 KB, cols n-major

  const int tid  = threadIdx.x;
  const int lane = tid & 31;
  const int wid  = tid >> 5;
  const int wy   = wid & 3;   // 0..3 -> 32-row slab
  const int wx   = wid >> 2;  // 0..1 -> 32-col slab
  const int row0 = blockIdx.x * 128;
  const int col0 = blockIdx.y * 64;

  v8f acc00 = {}, acc01 = {}, acc10 = {}, acc11 = {};

  for (int kt = 0; kt < K; kt += 32) {
    // ---- Stage A tile: raw bf16 copy, 8 elems (b128) per ld/st ----
#pragma unroll
    for (int w = tid; w < 128 * 4; w += 256) {   // 2 iters
      const int r = w >> 2, kq = w & 3;          // kq: 8-element group
      const int gm = row0 + r;
      const int gk = kt + (kq << 3);
      const __bf16* src;
      if (AMODE == A_XGATHER) {
        const int b = gm >> 11, t = gm & (TT - 1);
        const int s = gk >> 8, d = gk & 255;
        src = A + ((((long)(b * 3 + s) * TT + t) << 8) + d);
      } else if (AMODE == A_FLIP) {
        const int b = gm >> 11, t = gm & (TT - 1);
        src = A + (long)((b << 11) | (TT - 1 - t)) * lda + gk;
      } else {
        src = A + (long)gm * lda + gk;
      }
      *(uint4*)&As[r * STR + (kq << 2)] = *(const uint4*)src;
    }
    // ---- Stage B tile (32 x 64, k-pairs packed, stored n-major) ----
    if (BMODE == B_NORMAL) {
#pragma unroll
      for (int w = tid; w < 16 * 32; w += 256) {  // 2 iters
        const int kp = w >> 5;
        const int n2 = (w & 31) << 1;
        const int gk = kt + (kp << 1);
        const int gn = col0 + n2;
        unsigned int a = 0, b = 0;
        if (gn + 1 < N) {
          a = *(const unsigned int*)(B + (long)gk * ldb + gn);
          b = *(const unsigned int*)(B + (long)(gk + 1) * ldb + gn);
        } else if (gn < N) {
          a = (unsigned int)*(const unsigned short*)(B + (long)gk * ldb + gn);
          b = (unsigned int)*(const unsigned short*)(B + (long)(gk + 1) * ldb + gn);
        }
        Bs[n2 * STR + kp]       = (a & 0xffffu) | (b << 16);
        Bs[(n2 + 1) * STR + kp] = (a >> 16) | (b & 0xffff0000u);
      }
    } else {  // B_TRANS: k contiguous in memory -> single dword per k-pair
#pragma unroll
      for (int w = tid; w < 16 * 64; w += 256) {  // 4 iters
        const int n = w & 63, kp = w >> 6;
        const int gk = kt + (kp << 1);
        const int gn = col0 + n;
        unsigned int v = 0;
        if (gn < N) v = *(const unsigned int*)(B + (long)gn * ldb + gk);
        Bs[n * STR + kp] = v;
      }
    }
    // ---- Prefetch next K-tile into cache while WMMAs run ----
    if (kt + 32 < K) {
      const int pr = tid & 127;
      const __bf16* pa;
      if (AMODE == A_XGATHER) {
        const int gm = row0 + pr, gk = kt + 32;
        const int b = gm >> 11, t = gm & (TT - 1);
        const int s = gk >> 8, d = gk & 255;
        pa = A + ((((long)(b * 3 + s) * TT + t) << 8) + d);
      } else if (AMODE == A_FLIP) {
        const int gm = row0 + pr;
        const int b = gm >> 11, t = gm & (TT - 1);
        pa = A + (long)((b << 11) | (TT - 1 - t)) * lda + (kt + 32);
      } else {
        pa = A + (long)(row0 + pr) * lda + (kt + 32);
      }
      __builtin_prefetch(pa, 0, 1);
      if (tid < 32) {
        const __bf16* pb;
        if (BMODE == B_TRANS) pb = B + (long)(col0 + tid) * ldb + (kt + 32);
        else                  pb = B + (long)(kt + 32 + tid) * ldb + col0;
        __builtin_prefetch(pb, 0, 1);
      }
    }
    __syncthreads();

    const v16bf a0 = frag_ld(As, wy * 32, lane);
    const v16bf a1 = frag_ld(As, wy * 32 + 16, lane);
    const v16bf b0 = frag_ld(Bs, wx * 32, lane);
    const v16bf b1 = frag_ld(Bs, wx * 32 + 16, lane);
    acc00 = __builtin_amdgcn_wmma_f32_16x16x32_bf16(false, a0, false, b0, (short)0, acc00, false, false);
    acc01 = __builtin_amdgcn_wmma_f32_16x16x32_bf16(false, a0, false, b1, (short)0, acc01, false, false);
    acc10 = __builtin_amdgcn_wmma_f32_16x16x32_bf16(false, a1, false, b0, (short)0, acc10, false, false);
    acc11 = __builtin_amdgcn_wmma_f32_16x16x32_bf16(false, a1, false, b1, (short)0, acc11, false, false);
    __syncthreads();
  }

  // ---- Epilogue: C/D layout -> lane n=lane&15, m = +(lane>>4)*8 + r ----
  const int moff = row0 + wy * 32 + ((lane >> 4) << 3);
  const int noff = col0 + wx * 32 + (lane & 15);
#pragma unroll
  for (int mt = 0; mt < 2; ++mt) {
#pragma unroll
    for (int r = 0; r < 8; ++r) {
      const int gm = moff + mt * 16 + r;
      long orow;
      if (EPI == EPI_FLIPROW) {
        const int b = gm >> 11, t = gm & (TT - 1);
        orow = (long)((b << 11) | (TT - 1 - t)) * ldc;
      } else {
        orow = (long)gm * ldc;
      }
      const v8f* accs0 = (mt == 0) ? &acc00 : &acc10;
      const v8f* accs1 = (mt == 0) ? &acc01 : &acc11;
      const int n0 = noff, n1 = noff + 16;
      if (n0 < N) {
        float v = (*accs0)[r];
        if (EPI == EPI_BIAS) v += bias[n0];
        C[orow + n0] = v;
        if (DUAL) Cbf[orow + n0] = (__bf16)v;
      }
      if (n1 < N) {
        float v = (*accs1)[r];
        if (EPI == EPI_BIAS) v += bias[n1];
        C[orow + n1] = v;
        if (DUAL) Cbf[orow + n1] = (__bf16)v;
      }
    }
  }
}

// ---------------------------------------------------------------------------
// f32 -> bf16 conversion pass (2 elems/thread, dword stores)
// ---------------------------------------------------------------------------
__global__ __launch_bounds__(256) void cvt_bf16_kernel(
    const float* __restrict__ src, __bf16* __restrict__ dst, long n) {
  const long i = ((long)blockIdx.x * 256 + threadIdx.x) * 2;
  if (i + 1 < n) {
    const float2 v = *(const float2*)(src + i);
    *(unsigned int*)(dst + i) = pack_bf16x2(v.x, v.y);
  } else if (i < n) {
    dst[i] = (__bf16)src[i];
  }
}

// ---------------------------------------------------------------------------
// Block reduce (sum) helper
// ---------------------------------------------------------------------------
__device__ __forceinline__ float blockReduceSum(float v, float* sd) {
  const int tid = threadIdx.x;
  sd[tid] = v;
  __syncthreads();
#pragma unroll
  for (int off = 128; off > 0; off >>= 1) {
    if (tid < off) sd[tid] += sd[tid + off];
    __syncthreads();
  }
  const float r = sd[0];
  __syncthreads();
  return r;
}

// LayerNorm(pre) -> h (bf16 for GEMM), gate = silu(h) (f32).  Width 512.
__global__ __launch_bounds__(256) void ln_gate_kernel(
    const float* __restrict__ pre, const float* __restrict__ g,
    const float* __restrict__ bb, __bf16* __restrict__ hbf,
    float* __restrict__ gate) {
  __shared__ float sd[256];
  const long row = blockIdx.x;
  const float* r = pre + row * DM2;
  const int i0 = threadIdx.x * 2, i1 = i0 + 1;
  const float v0 = r[i0], v1 = r[i1];
  const float mu = blockReduceSum(v0 + v1, sd) * (1.0f / DM2);
  const float d0 = v0 - mu, d1 = v1 - mu;
  const float var = blockReduceSum(d0 * d0 + d1 * d1, sd) * (1.0f / DM2);
  const float rs = rsqrtf(var + 1e-5f);
  const float h0 = d0 * rs * g[i0] + bb[i0];
  const float h1 = d1 * rs * g[i1] + bb[i1];
  *(unsigned int*)(hbf + row * DM2 + i0) = pack_bf16x2(h0, h1);
  float2 gt;
  gt.x = siluf(h0);
  gt.y = siluf(h1);
  *(float2*)(gate + row * DM2 + i0) = gt;
}

// dt = softplus(zx[..., 2304+h] + dtb); dA = exp(dt * -exp(Alog))
__global__ __launch_bounds__(256) void dt_kernel(
    const float* __restrict__ zx, const float* __restrict__ dtbias,
    const float* __restrict__ Alog, float* __restrict__ dtv, float* __restrict__ dAv) {
  const long i = (long)blockIdx.x * blockDim.x + threadIdx.x;
  if (i >= (long)MROWS * NH) return;
  const long bt = i >> 4;
  const int hh = (int)(i & 15);
  const float x = zx[bt * DPRJ + (DPRJ - NH) + hh] + dtbias[hh];
  const float sp = (x > 20.0f) ? x : log1pf(__expf(x));
  const float A = -__expf(Alog[hh]);
  dtv[i] = sp;
  dAv[i] = __expf(sp * A);
}

// Depthwise causal conv(4) over xBC channels + SiLU
__global__ __launch_bounds__(256) void conv_kernel(
    const float* __restrict__ zx, const float* __restrict__ cw,
    const float* __restrict__ cb, float* __restrict__ xconv) {
  const long i = (long)blockIdx.x * blockDim.x + threadIdx.x;
  if (i >= (long)MROWS * CCH) return;
  const int c = (int)(i % CCH);
  const long bt = i / CCH;
  const int b = (int)(bt >> 11);
  const int t = (int)(bt & (TT - 1));
  float acc = cb[c];
#pragma unroll
  for (int k = 0; k < 4; ++k) {
    const int tt = t - 3 + k;
    if (tt >= 0) acc += cw[c * 4 + k] * zx[((long)(b << 11) + tt) * DPRJ + DIN + c];
  }
  xconv[i] = siluf(acc);
}

// Sequential selective scan. One block per (head, batch); state 64x128 f32 in
// registers (32 floats / thread).  y includes the Dh * x skip term.
__global__ __launch_bounds__(256) void scan_kernel(
    const float* __restrict__ xconv, const float* __restrict__ dtv,
    const float* __restrict__ dAv, const float* __restrict__ Dh,
    float* __restrict__ ybuf) {
  const int h = blockIdx.x;  // 0..15
  const int b = blockIdx.y;  // 0..3
  const int tid = threadIdx.x;
  const int p = tid & 63;
  const int j = tid >> 6;  // 0..3 -> n-chunk of 32
  __shared__ float xsh[64], Bsh[128], Csh[128], part[256], scal[2];
  float st[32];
#pragma unroll
  for (int k = 0; k < 32; ++k) st[k] = 0.0f;
  const float dh = Dh[h];

  for (int t = 0; t < TT; ++t) {
    const long bt = ((long)b << 11) + t;
    const float* xrow = xconv + bt * CCH;
    if (tid < 128) {
      Bsh[tid] = xrow[DIN + tid];
      Csh[tid] = xrow[DIN + DST + tid];
    } else if (tid < 192) {
      xsh[tid - 128] = xrow[(h << 6) + (tid - 128)];
    } else if (tid == 192) {
      const long ih = bt * NH + h;
      scal[0] = dAv[ih];
      scal[1] = dtv[ih];
    }
    __syncthreads();

    const float dA = scal[0];
    const float sc = scal[1] * xsh[p];
    float ps = 0.0f;
    const float* Bn = Bsh + j * 32;
    const float* Cn = Csh + j * 32;
#pragma unroll
    for (int k = 0; k < 32; ++k) {
      st[k] = dA * st[k] + sc * Bn[k];
      ps += st[k] * Cn[k];
    }
    part[tid] = ps;
    __syncthreads();
    if (tid < 64) {
      const float y = part[tid] + part[tid + 64] + part[tid + 128] + part[tid + 192] + dh * xsh[tid];
      ybuf[(bt * NH + h) * HD + tid] = y;
    }
    __syncthreads();
  }
}

// y = ys * silu(z); RMSNorm * normw -> bf16 for out-proj GEMM.  Width 1024.
__global__ __launch_bounds__(256) void gated_norm_kernel(
    const float* __restrict__ ybuf, const float* __restrict__ zx,
    const float* __restrict__ normw, __bf16* __restrict__ ygnbf) {
  __shared__ float sd[256];
  const long row = blockIdx.x;
  const float* yr = ybuf + row * DIN;
  const float* zr = zx + row * DPRJ;
  float v[4];
  float ss = 0.0f;
#pragma unroll
  for (int i = 0; i < 4; ++i) {
    const int c = threadIdx.x * 4 + i;
    const float y = yr[c] * siluf(zr[c]);
    v[i] = y;
    ss += y * y;
  }
  const float ms = blockReduceSum(ss, sd) * (1.0f / DIN);
  const float sc = rsqrtf(ms + 1e-5f);
  const int c0 = threadIdx.x * 4;
  uint2 pk;
  pk.x = pack_bf16x2(v[0] * sc * normw[c0],     v[1] * sc * normw[c0 + 1]);
  pk.y = pack_bf16x2(v[2] * sc * normw[c0 + 2], v[3] * sc * normw[c0 + 3]);
  *(uint2*)(ygnbf + row * DIN + c0) = pk;
}

// tmp = (out_f + out_b) * gate -> bf16 (2 elems/thread)
__global__ __launch_bounds__(256) void combine_kernel(
    const float* __restrict__ of, const float* __restrict__ ob,
    const float* __restrict__ gate, __bf16* __restrict__ tmpbf, long n) {
  const long i = ((long)blockIdx.x * 256 + threadIdx.x) * 2;
  if (i + 1 < n) {
    const float2 a = *(const float2*)(of + i);
    const float2 b = *(const float2*)(ob + i);
    const float2 g = *(const float2*)(gate + i);
    *(unsigned int*)(tmpbf + i) = pack_bf16x2((a.x + b.x) * g.x, (a.y + b.y) * g.y);
  }
}

// LayerNorm(post) + shortcut -> out; fn = out / max(||out||, 1e-12) -> bf16
__global__ __launch_bounds__(256) void lnpost_fn_kernel(
    const float* __restrict__ post, const float* __restrict__ shortcut,
    const float* __restrict__ g, const float* __restrict__ bb,
    float* __restrict__ out, __bf16* __restrict__ fnbf) {
  __shared__ float sd[256];
  const long row = blockIdx.x;
  const int i = threadIdx.x;
  const float v = post[row * DMOD + i];
  const float mu = blockReduceSum(v, sd) * (1.0f / DMOD);
  const float d = v - mu;
  const float var = blockReduceSum(d * d, sd) * (1.0f / DMOD);
  const float rs = rsqrtf(var + 1e-5f);
  const float o = d * rs * g[i] + bb[i] + shortcut[row * DMOD + i];
  out[row * DMOD + i] = o;
  const float nn = blockReduceSum(o * o, sd);
  const float inv = 1.0f / fmaxf(sqrtf(nn), 1e-12f);
  fnbf[row * DMOD + i] = (__bf16)(o * inv);
}

// ---------------------------------------------------------------------------
extern "C" void kernel_launch(void* const* d_in, const int* in_sizes, int n_in,
                              void* d_out, int out_size, void* d_ws, size_t ws_size,
                              hipStream_t stream) {
  (void)in_sizes; (void)n_in; (void)out_size; (void)ws_size;
  const float* x       = (const float*)d_in[0];
  const float* Wp      = (const float*)d_in[1];
  const float* bp      = (const float*)d_in[2];
  const float* Wpre    = (const float*)d_in[3];
  const float* bpre    = (const float*)d_in[4];
  const float* lnpre_g = (const float*)d_in[5];
  const float* lnpre_b = (const float*)d_in[6];
  const float* Wpost   = (const float*)d_in[7];
  const float* bpost   = (const float*)d_in[8];
  const float* lnpost_g = (const float*)d_in[9];
  const float* lnpost_b = (const float*)d_in[10];
  const float* prm[2][8];
  for (int dir = 0; dir < 2; ++dir)
    for (int k = 0; k < 8; ++k) prm[dir][k] = (const float*)d_in[11 + dir * 8 + k];
  // prm[dir]: 0=Win 1=convw 2=convb 3=dtb 4=Alog 5=Dh 6=normw 7=Wout

  float* outmain = (float*)d_out;
  float* attn    = outmain + (long)MROWS * DMOD;

  char* wsb = (char*)d_ws;
  size_t off = 0;
  auto allocf = [&](size_t n) {
    off = (off + 15) & ~(size_t)15;
    float* p = (float*)(wsb + off);
    off += n * 4;
    return p;
  };
  auto allocb = [&](size_t n) {
    off = (off + 15) & ~(size_t)15;
    __bf16* p = (__bf16*)(wsb + off);
    off += n * 2;
    return p;
  };
  // f32 intermediates
  float* shortcut = allocf((size_t)MROWS * DMOD);
  float* pre      = allocf((size_t)MROWS * DM2);
  float* gate     = allocf((size_t)MROWS * DM2);
  float* zx       = allocf((size_t)MROWS * DPRJ);
  float* xconv    = allocf((size_t)MROWS * CCH);
  float* dtv      = allocf((size_t)MROWS * NH);
  float* dAv      = allocf((size_t)MROWS * NH);
  float* ybuf     = allocf((size_t)MROWS * DIN);
  float* out_f    = allocf((size_t)MROWS * DM2);
  float* out_b    = allocf((size_t)MROWS * DM2);
  float* postb    = allocf((size_t)MROWS * DMOD);
  // bf16 GEMM operands
  __bf16* x_bf     = allocb((size_t)BB * 3 * TT * DMOD);
  __bf16* wp_bf    = allocb((size_t)3 * DMOD * DMOD);
  __bf16* wpre_bf  = allocb((size_t)DMOD * DM2);
  __bf16* wpost_bf = allocb((size_t)DM2 * DMOD);
  __bf16* win_bf[2]  = {allocb((size_t)DM2 * DPRJ), allocb((size_t)DM2 * DPRJ)};
  __bf16* wout_bf[2] = {allocb((size_t)DIN * DM2), allocb((size_t)DIN * DM2)};
  __bf16* sc_bf  = allocb((size_t)MROWS * DMOD);
  __bf16* h_bf   = allocb((size_t)MROWS * DM2);
  __bf16* ygn_bf = allocb((size_t)MROWS * DIN);
  __bf16* tmp_bf = allocb((size_t)MROWS * DM2);
  __bf16* fn_bf  = allocb((size_t)MROWS * DMOD);

  const dim3 blk(256);
  auto cvt = [&](const float* s, __bf16* d, long n) {
    cvt_bf16_kernel<<<(unsigned)((n / 2 + 255) / 256), blk, 0, stream>>>(s, d, n);
  };
  // 0) one-time f32 -> bf16 conversions of GEMM operands
  cvt(x, x_bf, (long)BB * 3 * TT * DMOD);
  cvt(Wp, wp_bf, (long)3 * DMOD * DMOD);
  cvt(Wpre, wpre_bf, (long)DMOD * DM2);
  cvt(Wpost, wpost_bf, (long)DM2 * DMOD);
  for (int dir = 0; dir < 2; ++dir) {
    cvt(prm[dir][0], win_bf[dir], (long)DM2 * DPRJ);
    cvt(prm[dir][7], wout_bf[dir], (long)DIN * DM2);
  }

  // 1) shortcut = gather(x) @ Wp + bp   (8192 x 256, K=768), dual f32+bf16
  gemm_wmma<A_XGATHER, B_NORMAL, EPI_BIAS, 1><<<dim3(MROWS / 128, 4, 1), blk, 0, stream>>>(
      x_bf, wp_bf, bp, shortcut, sc_bf, MROWS, DMOD, 3 * DMOD, 0, DMOD, DMOD, 0, 0, 0);
  // 2) pre = shortcut @ Wpre + bpre     (8192 x 512, K=256)
  gemm_wmma<A_NORMAL, B_NORMAL, EPI_BIAS, 0><<<dim3(MROWS / 128, 8, 1), blk, 0, stream>>>(
      sc_bf, wpre_bf, bpre, pre, nullptr, MROWS, DM2, DMOD, DMOD, DM2, DM2, 0, 0, 0);
  // 3) layernorm -> h (bf16), gate (f32)
  ln_gate_kernel<<<MROWS, blk, 0, stream>>>(pre, lnpre_g, lnpre_b, h_bf, gate);

  auto run_dir = [&](int dir, float* outp) {
    // in-proj (8192 x 2320, K=512); backward reads time-flipped h
    if (dir == 1)
      gemm_wmma<A_FLIP, B_NORMAL, EPI_STORE, 0><<<dim3(MROWS / 128, (DPRJ + 63) / 64, 1), blk, 0, stream>>>(
          h_bf, win_bf[dir], nullptr, zx, nullptr, MROWS, DPRJ, DM2, DM2, DPRJ, DPRJ, 0, 0, 0);
    else
      gemm_wmma<A_NORMAL, B_NORMAL, EPI_STORE, 0><<<dim3(MROWS / 128, (DPRJ + 63) / 64, 1), blk, 0, stream>>>(
          h_bf, win_bf[dir], nullptr, zx, nullptr, MROWS, DPRJ, DM2, DM2, DPRJ, DPRJ, 0, 0, 0);
    dt_kernel<<<(MROWS * NH + 255) / 256, blk, 0, stream>>>(zx, prm[dir][3], prm[dir][4], dtv, dAv);
    conv_kernel<<<((long)MROWS * CCH + 255) / 256, blk, 0, stream>>>(zx, prm[dir][1], prm[dir][2], xconv);
    scan_kernel<<<dim3(NH, BB), blk, 0, stream>>>(xconv, dtv, dAv, prm[dir][5], ybuf);
    gated_norm_kernel<<<MROWS, blk, 0, stream>>>(ybuf, zx, prm[dir][6], ygn_bf);
    // out-proj (8192 x 512, K=1024); backward writes time-flipped rows
    if (dir == 1)
      gemm_wmma<A_NORMAL, B_NORMAL, EPI_FLIPROW, 0><<<dim3(MROWS / 128, 8, 1), blk, 0, stream>>>(
          ygn_bf, wout_bf[dir], nullptr, outp, nullptr, MROWS, DM2, DIN, DIN, DM2, DM2, 0, 0, 0);
    else
      gemm_wmma<A_NORMAL, B_NORMAL, EPI_STORE, 0><<<dim3(MROWS / 128, 8, 1), blk, 0, stream>>>(
          ygn_bf, wout_bf[dir], nullptr, outp, nullptr, MROWS, DM2, DIN, DIN, DM2, DM2, 0, 0, 0);
  };
  run_dir(0, out_f);
  run_dir(1, out_b);

  // combine: tmp = (out_f + out_b) * gate  -> bf16
  combine_kernel<<<((long)MROWS * DM2 / 2 + 255) / 256, blk, 0, stream>>>(
      out_f, out_b, gate, tmp_bf, (long)MROWS * DM2);
  // post-proj (8192 x 256, K=512)
  gemm_wmma<A_NORMAL, B_NORMAL, EPI_BIAS, 0><<<dim3(MROWS / 128, 4, 1), blk, 0, stream>>>(
      tmp_bf, wpost_bf, bpost, postb, nullptr, MROWS, DMOD, DM2, DM2, DMOD, DMOD, 0, 0, 0);
  // layernorm + residual -> out (f32); fn (bf16 row-normalized)
  lnpost_fn_kernel<<<MROWS, blk, 0, stream>>>(postb, shortcut, lnpost_g, lnpost_b, outmain, fn_bf);
  // attn[b] = fn[b] @ fn[b]^T  (batched 2048 x 2048, K=256)
  gemm_wmma<A_NORMAL, B_TRANS, EPI_STORE, 0><<<dim3(TT / 128, TT / 64, BB), blk, 0, stream>>>(
      fn_bf, fn_bf, nullptr, attn, nullptr, TT, TT, DMOD, DMOD, DMOD, TT,
      (long)TT * DMOD, (long)TT * DMOD, (long)TT * TT);
}
